// HeteroGNN_35562329210980
// MI455X (gfx1250) — matrix operand order repaired and checked by reading, not compiled, not run on hardware
//
#include <hip/hip_runtime.h>

// HeteroGNN (PyG SAGEConv mean) on MI455X / gfx1250.
// Only the subject->subject chain is live w.r.t. the output; the roi branch
// (sr/rr edges, x_roi, all *_sr/*_rr params) is dead code and skipped.

constexpr int DF = 128;   // feature dim D == hidden H

typedef __attribute__((ext_vector_type(16))) __bf16 v16bf;
typedef __attribute__((ext_vector_type(8)))  float  v8f;

// B-fragment-packed weights: [kc(8)][ntile(8)][lane(32)][t(16)] bf16
constexpr int BPK_ELEMS = 8 * 8 * 32 * 16;   // 32768 bf16 = 64 KB per layer

// ---------------------------------------------------------------------------
// Scatter-add of 128-float rows along edges: acc[dst] += feat[src]; cnt[dst]++
// One wave (32 lanes) per edge; each lane handles 4 coalesced floats.
// ---------------------------------------------------------------------------
__global__ void scatter_kernel(const float* __restrict__ feat,
                               const int*   __restrict__ src,
                               const int*   __restrict__ dst,
                               int nE,
                               float* __restrict__ acc,
                               float* __restrict__ cnt) {
    const int  lane = threadIdx.x & 31;
    const long wave = (long)blockIdx.x * (blockDim.x >> 5) + (threadIdx.x >> 5);
    const long nw   = (long)gridDim.x * (blockDim.x >> 5);
    for (long e = wave; e < nE; e += nw) {
        const int s = src[e];
        const int d = dst[e];
        const float* srow = feat + (size_t)s * DF;
        float*       drow = acc  + (size_t)d * DF;
#pragma unroll
        for (int t = 0; t < 4; ++t) {
            const int f = lane + 32 * t;
            unsafeAtomicAdd(&drow[f], srow[f]);
        }
        if (cnt != nullptr && lane == 0) unsafeAtomicAdd(&cnt[d], 1.0f);
    }
}

// ---------------------------------------------------------------------------
// acc[i][f] /= max(cnt[i], 1)   (mean aggregation)
// ---------------------------------------------------------------------------
__global__ void normalize_kernel(float* __restrict__ acc,
                                 const float* __restrict__ cnt,
                                 int rows) {
    const size_t i = (size_t)blockIdx.x * blockDim.x + threadIdx.x;
    if (i >= (size_t)rows * DF) return;
    const int r = (int)(i >> 7);               // / 128
    acc[i] = acc[i] / fmaxf(cnt[r], 1.0f);
}

// ---------------------------------------------------------------------------
// Pack one layer's weights [W1 ; W2] (each 128x128 f32, K-major rows) into the
// exact WMMA B-fragment layout as bf16:
//   out[((kc*8 + w)*32 + lane)*16 + t] = W[k][col]
//   k    = (kc*32 + 16*(lane>>4) + t) mod 128     (kc<4 -> W1, else W2)
//   col  = w*16 + (lane&15)
// ---------------------------------------------------------------------------
__global__ void pack_weights_kernel(const float* __restrict__ W1,
                                    const float* __restrict__ W2,
                                    __bf16* __restrict__ out) {
    const int idx = blockIdx.x * blockDim.x + threadIdx.x;
    if (idx >= BPK_ELEMS) return;
    const int t    = idx & 15;
    const int lane = (idx >> 4) & 31;
    const int w    = (idx >> 9) & 7;
    const int kc   = (idx >> 12) & 7;
    const int k    = kc * 32 + 16 * (lane >> 4) + t;   // concat-K in [0,256)
    const int col  = w * 16 + (lane & 15);
    const float* W = (kc < 4) ? W1 : W2;
    const int kw   = (kc < 4) ? k : k - DF;
    out[idx] = (__bf16)W[(size_t)kw * DF + col];
}

// ---------------------------------------------------------------------------
// Fused SAGE layer: Out = relu(A1 @ W1 + A2 @ W2 + bias)
//   A1 = mean-aggregated neighbors (M x 128), A2 = self features (M x 128)
//   Bpk = packed bf16 weights for [W1 ; W2] in B-fragment order.
// Block = 16-row strip; 8 waves, wave w owns 16-col tile w.
// bf16 WMMA 16x16x32, f32 accumulate. M must be a multiple of 16 (50000 = 3125*16).
// ---------------------------------------------------------------------------
__global__ __launch_bounds__(256)
void sage_gemm_kernel(const float* __restrict__ A1,
                      const float* __restrict__ A2,
                      const __bf16* __restrict__ Bpk,
                      const float* __restrict__ bias,
                      float* __restrict__ Out,
                      int M) {
    __shared__ __bf16 sA[16][2 * DF + 8];      // [row][concat K], padded

    const int tid  = threadIdx.x;
    const int row0 = blockIdx.x * 16;

    // Stage the 16x256 concat strip [A1 | A2] into LDS as bf16.
    // A 16-row strip of a row-major Mx128 matrix is one contiguous 8 KB block,
    // so read with float4 (global_load_b128) and store packed bf16 quads.
    {
        const float4* A1v = (const float4*)(A1 + (size_t)row0 * DF);
        const float4* A2v = (const float4*)(A2 + (size_t)row0 * DF);
#pragma unroll
        for (int pass = 0; pass < 2; ++pass) {          // 512 float4 / 256 thr
            const int idx = tid + pass * 256;
            const int r  = idx >> 5;                    // 32 float4 per row
            const int c4 = (idx & 31) * 4;
            const float4 v1 = A1v[idx];
            const float4 v2 = A2v[idx];
            sA[r][c4 + 0] = (__bf16)v1.x;  sA[r][c4 + 1] = (__bf16)v1.y;
            sA[r][c4 + 2] = (__bf16)v1.z;  sA[r][c4 + 3] = (__bf16)v1.w;
            sA[r][DF + c4 + 0] = (__bf16)v2.x;  sA[r][DF + c4 + 1] = (__bf16)v2.y;
            sA[r][DF + c4 + 2] = (__bf16)v2.z;  sA[r][DF + c4 + 3] = (__bf16)v2.w;
        }
    }
    __syncthreads();

    const int lane = tid & 31;
    const int w    = tid >> 5;                 // N-tile index 0..7
    const int n    = lane & 15;                // col within tile / A row
    const int hi   = lane >> 4;                // lane half
    const int col  = w * 16 + n;

    v8f c = {};
#pragma unroll
    for (int kc = 0; kc < 8; ++kc) {
        const int k0 = kc * 32;                // offset in concat-K space [0,256)

        // A fragment (16x32 bf16): lane holds row n; K = 16*hi+t (contiguous)
        v16bf a;
#pragma unroll
        for (int t = 0; t < 8; ++t) {
            a[t]     = sA[n][k0 + 16 * hi + t];
            a[8 + t] = sA[n][k0 + 16 * hi + 8 + t];
        }
        // B fragment: contiguous 32 B per lane from the packed weight image.
        const v16bf b = *(const v16bf*)(Bpk + (((size_t)(kc * 8 + w) * 32 + lane) << 4));

        c = __builtin_amdgcn_wmma_f32_16x16x32_bf16(false, a, false, b,
                                                    (short)0, c, false, false);
    }

    const float bv = bias[col];
#pragma unroll
    for (int j = 0; j < 8; ++j) {
        float v = c[j] + bv;
        v = fmaxf(v, 0.0f);                    // relu
        Out[(size_t)(row0 + j + 8 * hi) * DF + col] = v;
    }
}

// ---------------------------------------------------------------------------
// out[i][o] = sum_f X[i][f] * Wl[f][o] + bl[o],  O = 2.
// One wave per row: coalesced feature reads, shuffle-xor reduction.
// ---------------------------------------------------------------------------
__global__ void final_linear_kernel(const float* __restrict__ X,
                                    const float* __restrict__ Wl,
                                    const float* __restrict__ bl,
                                    float* __restrict__ out,
                                    int M) {
    const int  lane = threadIdx.x & 31;
    const long row  = (long)blockIdx.x * (blockDim.x >> 5) + (threadIdx.x >> 5);
    if (row >= M) return;
    const float* xr = X + (size_t)row * DF;
    float a0 = 0.0f, a1 = 0.0f;
#pragma unroll
    for (int t = 0; t < 4; ++t) {
        const int f = lane + 32 * t;
        const float v = xr[f];
        a0 += v * Wl[f * 2 + 0];
        a1 += v * Wl[f * 2 + 1];
    }
#pragma unroll
    for (int off = 16; off > 0; off >>= 1) {
        a0 += __shfl_xor(a0, off, 32);
        a1 += __shfl_xor(a1, off, 32);
    }
    if (lane == 0) {
        out[row * 2 + 0] = a0 + bl[0];
        out[row * 2 + 1] = a1 + bl[1];
    }
}

// ---------------------------------------------------------------------------
extern "C" void kernel_launch(void* const* d_in, const int* in_sizes, int n_in,
                              void* d_out, int out_size, void* d_ws, size_t ws_size,
                              hipStream_t stream) {
    // setup_inputs() order
    const float* x_subject = (const float*)d_in[0];
    const float* w1_ss_l   = (const float*)d_in[6];
    const float* w1_ss_r   = (const float*)d_in[7];
    const float* w2_ss_l   = (const float*)d_in[12];
    const float* w2_ss_r   = (const float*)d_in[13];
    const float* b1_ss     = (const float*)d_in[16];
    const float* b2_ss     = (const float*)d_in[19];
    const float* lin_w     = (const float*)d_in[20];
    const float* lin_b     = (const float*)d_in[21];
    const int*   ss_src    = (const int*)d_in[26];
    const int*   ss_dst    = (const int*)d_in[27];

    const int NS  = in_sizes[0] / DF;          // 50000
    const int ESS = in_sizes[26];              // 800000

    // Workspace layout: acc | s1 | s2 | cnt (f32), then packed weights (bf16).
    float* acc = (float*)d_ws;
    float* s1  = acc + (size_t)NS * DF;
    float* s2  = s1  + (size_t)NS * DF;
    float* cnt = s2  + (size_t)NS * DF;
    size_t bpkOff = ((size_t)NS * DF * 3 + NS) * sizeof(float);
    bpkOff = (bpkOff + 63) & ~(size_t)63;
    __bf16* Bpk1 = (__bf16*)((char*)d_ws + bpkOff);
    __bf16* Bpk2 = Bpk1 + BPK_ELEMS;

    const size_t featBytes = (size_t)NS * DF * sizeof(float);
    const int sc_blocks   = (ESS + 7) / 8;                       // 8 waves/block
    const int nrm_blocks  = (int)(((size_t)NS * DF + 255) / 256);
    const int gemm_blocks = NS / 16;                             // 3125
    const int fin_blocks  = (NS + 7) / 8;
    const int pk_blocks   = (BPK_ELEMS + 255) / 256;

    // Pack weights into B-fragment layout (tiny; overlaps with scatter).
    pack_weights_kernel<<<pk_blocks, 256, 0, stream>>>(w1_ss_l, w1_ss_r, Bpk1);
    pack_weights_kernel<<<pk_blocks, 256, 0, stream>>>(w2_ss_l, w2_ss_r, Bpk2);

    // ----- layer 1 (subject branch only; roi branch is dead) -----
    hipMemsetAsync(acc, 0, featBytes, stream);
    hipMemsetAsync(cnt, 0, (size_t)NS * sizeof(float), stream);
    scatter_kernel<<<sc_blocks, 256, 0, stream>>>(x_subject, ss_src, ss_dst, ESS, acc, cnt);
    normalize_kernel<<<nrm_blocks, 256, 0, stream>>>(acc, cnt, NS);
    sage_gemm_kernel<<<gemm_blocks, 256, 0, stream>>>(acc, x_subject, Bpk1, b1_ss, s1, NS);

    // ----- layer 2 (same edge set -> reuse counts) -----
    hipMemsetAsync(acc, 0, featBytes, stream);
    scatter_kernel<<<sc_blocks, 256, 0, stream>>>(s1, ss_src, ss_dst, ESS, acc, nullptr);
    normalize_kernel<<<nrm_blocks, 256, 0, stream>>>(acc, cnt, NS);
    sage_gemm_kernel<<<gemm_blocks, 256, 0, stream>>>(acc, s1, Bpk2, b2_ss, s2, NS);

    // ----- output projection -----
    final_linear_kernel<<<fin_blocks, 256, 0, stream>>>(s2, lin_w, lin_b,
                                                        (float*)d_out, NS);
}
